// FSQuantizer_85564338471365
// MI455X (gfx1250) — matrix operand order repaired and checked by reading, not compiled, not run on hardware
//
#include <hip/hip_runtime.h>
#include <hip/hip_bf16.h>
#include <stdint.h>
#include <stddef.h>

// Problem constants (match reference)
#define DIMV   256
#define KCODES 4096
#define BATCH  8
#define NSEQ   2048
#define NROWS  (BATCH * NSEQ)   // 16384 flat x rows
#define COMMIT_W 0.25f

#ifndef __has_builtin
#define __has_builtin(x) 0
#endif

typedef __attribute__((ext_vector_type(4)))  int           v4i;
typedef __attribute__((ext_vector_type(16))) __bf16        v16bf;
typedef __attribute__((ext_vector_type(8)))  float         v8f;
typedef __attribute__((ext_vector_type(8)))  unsigned int  v8u;

#if __has_builtin(__builtin_amdgcn_global_load_async_to_lds_b128)
#define USE_ASYNC_COPY 1
typedef __attribute__((address_space(1))) v4i gv4i;  // global int4
typedef __attribute__((address_space(3))) v4i lv4i;  // LDS int4
#else
#define USE_ASYNC_COPY 0
#endif

__device__ __forceinline__ unsigned short f2bf_rne(float f) {
    unsigned int u = __float_as_uint(f);
    u += 0x7FFFu + ((u >> 16) & 1u);
    return (unsigned short)(u >> 16);
}
__device__ __forceinline__ float bf2f(unsigned short h) {
    return __uint_as_float(((unsigned int)h) << 16);
}

// ---------------------------------------------------------------------------
// Prep: codebook -> bf16 hi/lo planes + fp32 row norms. One block per code.
// ---------------------------------------------------------------------------
__global__ void __launch_bounds__(256)
prep_codebook(const float* __restrict__ cb,
              unsigned short* __restrict__ chi,
              unsigned short* __restrict__ clo,
              float* __restrict__ cnorm) {
    __shared__ float red[256];
    const int row = blockIdx.x;
    const int d   = threadIdx.x;
    const float v = cb[(size_t)row * DIMV + d];
    const unsigned short h = f2bf_rne(v);
    chi[(size_t)row * DIMV + d] = h;
    clo[(size_t)row * DIMV + d] = f2bf_rne(v - bf2f(h));
    red[d] = v * v;
    __syncthreads();
    for (int s = 128; s > 0; s >>= 1) {
        if (d < s) red[d] += red[d + s];
        __syncthreads();
    }
    if (d == 0) cnorm[row] = red[0];
}

// ---------------------------------------------------------------------------
// Prep: ||x_row||^2 with coalesced reads (x is [B, DIM, N], n fastest).
// ---------------------------------------------------------------------------
__global__ void __launch_bounds__(256)
prep_xnorm(const float* __restrict__ x, float* __restrict__ xnorm) {
    const int t = blockIdx.x * 256 + threadIdx.x;   // row = b*NSEQ + n
    const int b = t / NSEQ;
    const int n = t % NSEQ;
    const float* px = x + (size_t)b * DIMV * NSEQ + n;
    float acc = 0.f;
#pragma unroll 4
    for (int d = 0; d < DIMV; ++d) {
        const float v = px[(size_t)d * NSEQ];
        acc += v * v;
    }
    xnorm[t] = acc;
}

// ---------------------------------------------------------------------------
// Tile staging: 16 codes x 256 dims, hi & lo bf16 planes (8 KB each).
// Async path: GLOBAL_LOAD_ASYNC_TO_LDS_B128, no VGPR round-trip (ASYNCcnt).
// ---------------------------------------------------------------------------
__device__ __forceinline__ void
stage_tile(unsigned int* dhi, unsigned int* dlo,
           const unsigned int* ghi, const unsigned int* glo, int tid) {
#if USE_ASYNC_COPY
#pragma unroll
    for (int j = 0; j < 2; ++j) {
        const int c = (tid + 256 * j) * 4;  // uint index of 16B chunk
        __builtin_amdgcn_global_load_async_to_lds_b128(
            (gv4i*)(ghi + c), (lv4i*)(dhi + c), 0, 0);
        __builtin_amdgcn_global_load_async_to_lds_b128(
            (gv4i*)(glo + c), (lv4i*)(dlo + c), 0, 0);
    }
#else
#pragma unroll
    for (int j = 0; j < 8; ++j) {
        dhi[tid + 256 * j] = ghi[tid + 256 * j];
        dlo[tid + 256 * j] = glo[tid + 256 * j];
    }
#endif
}

// ---------------------------------------------------------------------------
// Main: bf16x3 WMMA distance GEMM + per-row argmin.
// Block = 256 threads = 8 waves; wave w owns M-tile (blockIdx.x*8 + w).
// Double-buffered async LDS staging of 16-code N-tiles; 3 independent WMMA
// accumulator chains (hi*hi, hi*lo, lo*hi) for XDL pipelining.
// ---------------------------------------------------------------------------
__global__ void __launch_bounds__(256)
argmin_wmma(const float* __restrict__ x,
            const unsigned short* __restrict__ chi,
            const unsigned short* __restrict__ clo,
            const float* __restrict__ cnorm,
            const float* __restrict__ xnorm,
            int*   __restrict__ out_idx,
            float* __restrict__ out_mind) {
    __shared__ __align__(32) unsigned int sh_hi[2][16 * DIMV / 2]; // 2 x 8 KB
    __shared__ __align__(32) unsigned int sh_lo[2][16 * DIMV / 2]; // 2 x 8 KB

    const int tid  = threadIdx.x;
    const int wave = tid >> 5;
    const int lane = tid & 31;
    const int mt   = blockIdx.x * 8 + wave;  // M-tile id, 0..1023
    const int g    = lane >> 4;              // half-wave group
    const int lr   = lane & 15;

    // ---- Build A fragments (hi & lo) for this wave's 16 x-rows. ----------
    // A layout (matches B): lanes 0-15 hold M=lane with K 0..15;
    // lanes 16-31 hold M=lane-16 with K 16..31 per 32-K slice.
    const int arow = mt * 16 + lr;
    const int ab   = arow / NSEQ;
    const int an   = arow % NSEQ;
    const float* px = x + (size_t)ab * DIMV * NSEQ + an;  // stride NSEQ per dim

    v8u a_hi[8], a_lo[8];
#pragma unroll
    for (int s = 0; s < 8; ++s) {
#pragma unroll
        for (int j = 0; j < 8; ++j) {
            const int e0 = 2 * j, e1 = 2 * j + 1;
            const int k0 = s * 32 + g * 16 + e0;
            const int k1 = s * 32 + g * 16 + e1;
            const float f0 = px[(size_t)k0 * NSEQ];
            const float f1 = px[(size_t)k1 * NSEQ];
            const unsigned short h0 = f2bf_rne(f0), h1 = f2bf_rne(f1);
            const unsigned short l0 = f2bf_rne(f0 - bf2f(h0));
            const unsigned short l1 = f2bf_rne(f1 - bf2f(h1));
            a_hi[s][j] = (unsigned int)h0 | ((unsigned int)h1 << 16);
            a_lo[s][j] = (unsigned int)l0 | ((unsigned int)l1 << 16);
        }
    }

    // C/D layout: lanes 0-15 -> rows 0..7 in v[0..7]; lanes 16-31 -> rows 8..15
    float xn[8];
#pragma unroll
    for (int r = 0; r < 8; ++r) xn[r] = xnorm[mt * 16 + g * 8 + r];

    float rmin[8];
    int   ridx[8];
#pragma unroll
    for (int r = 0; r < 8; ++r) { rmin[r] = 3.4e38f; ridx[r] = 0; }

    // Prime double buffer with tile 0.
    stage_tile(&sh_hi[0][0], &sh_lo[0][0],
               (const unsigned int*)chi, (const unsigned int*)clo, tid);

    for (int nt = 0; nt < KCODES / 16; ++nt) {
        const int cur = nt & 1;

#if USE_ASYNC_COPY
        asm volatile("s_wait_asynccnt 0x0" ::: "memory");
#endif
        __syncthreads();   // tile `cur` fully resident; prior reads of `cur^1` done

        if (nt + 1 < KCODES / 16) {
            const size_t off = (size_t)(nt + 1) * 16 * DIMV / 2; // uints
            stage_tile(&sh_hi[cur ^ 1][0], &sh_lo[cur ^ 1][0],
                       (const unsigned int*)chi + off,
                       (const unsigned int*)clo + off, tid);
        }

        // ---- bf16x3 GEMM over K=256: three independent accumulator chains
        v8f ahh = {}, ahl = {}, alh = {};
#pragma unroll
        for (int s = 0; s < 8; ++s) {
            // B layout used here (contiguous k, same as A): lanes 0-15 N=lane K 0..15; lanes 16-31 K 16..31
            const int koff = s * 32 + g * 16;
            const v8u bh = *(const v8u*)&sh_hi[cur][(lr * DIMV + koff) >> 1];
            const v8u bl = *(const v8u*)&sh_lo[cur][(lr * DIMV + koff) >> 1];
            const v16bf ah  = __builtin_bit_cast(v16bf, a_hi[s]);
            const v16bf al  = __builtin_bit_cast(v16bf, a_lo[s]);
            const v16bf bhv = __builtin_bit_cast(v16bf, bh);
            const v16bf blv = __builtin_bit_cast(v16bf, bl);
            ahh = __builtin_amdgcn_wmma_f32_16x16x32_bf16(
                false, ah, false, bhv, (short)0, ahh, false, false);
            ahl = __builtin_amdgcn_wmma_f32_16x16x32_bf16(
                false, ah, false, blv, (short)0, ahl, false, false);
            alh = __builtin_amdgcn_wmma_f32_16x16x32_bf16(
                false, al, false, bhv, (short)0, alh, false, false);
        }

        // ---- distances + running argmin (this lane's column) ----
        const int   ncol = nt * 16 + lr;
        const float cn   = cnorm[ncol];
#pragma unroll
        for (int r = 0; r < 8; ++r) {
            const float dot = ahh[r] + ahl[r] + alh[r];
            const float d   = xn[r] + cn - 2.0f * dot;
            if (d < rmin[r]) { rmin[r] = d; ridx[r] = ncol; }
        }
    }

    // ---- Reduce min over the 16 columns (lanes) of each half-wave ----
#pragma unroll
    for (int r = 0; r < 8; ++r) {
        float d = rmin[r];
        int   i = ridx[r];
#pragma unroll
        for (int m = 1; m < 16; m <<= 1) {
            const float od = __shfl_xor(d, m, 32);
            const int   oi = __shfl_xor(i, m, 32);
            if (od < d || (od == d && oi < i)) { d = od; i = oi; }
        }
        if (lr == 0) {
            const int row = mt * 16 + g * 8 + r;
            out_idx[row]  = i;
            out_mind[row] = d;   // == ||x - q||^2 (bf16x3-accurate)
        }
    }
}

// ---------------------------------------------------------------------------
// Gather quantized output back to [B, DIM, N] layout.
// ---------------------------------------------------------------------------
__global__ void __launch_bounds__(256)
write_quant(const float* __restrict__ cb, const int* __restrict__ idx,
            float* __restrict__ out) {
    const size_t t = (size_t)blockIdx.x * 256 + threadIdx.x;
    const int n = (int)(t % NSEQ);
    const int d = (int)((t / NSEQ) % DIMV);
    const int b = (int)(t / ((size_t)NSEQ * DIMV));
    const int code = idx[b * NSEQ + n];
    out[t] = cb[(size_t)code * DIMV + d];
}

__global__ void __launch_bounds__(256)
write_idx(const int* __restrict__ idx, float* __restrict__ out) {
    const int t = blockIdx.x * 256 + threadIdx.x;
    out[t] = (float)idx[t];
}

// Deterministic tree reduction of min-distances -> commit loss scalar.
__global__ void __launch_bounds__(256)
loss_reduce(const float* __restrict__ mind, float* __restrict__ out) {
    __shared__ float red[256];
    const int t = threadIdx.x;
    float s = 0.f;
    for (int j = 0; j < NROWS / 256; ++j) s += mind[t + 256 * j];
    red[t] = s;
    __syncthreads();
    for (int k = 128; k > 0; k >>= 1) {
        if (t < k) red[t] += red[t + k];
        __syncthreads();
    }
    if (t == 0) out[0] = red[0] * (COMMIT_W / ((float)NROWS * (float)DIMV));
}

// ---------------------------------------------------------------------------
extern "C" void kernel_launch(void* const* d_in, const int* in_sizes, int n_in,
                              void* d_out, int out_size, void* d_ws, size_t ws_size,
                              hipStream_t stream) {
    (void)in_sizes; (void)n_in; (void)out_size; (void)ws_size;
    const float* x  = (const float*)d_in[0];  // [B, DIM, N]
    const float* cb = (const float*)d_in[1];  // [K, DIM]
    float* out = (float*)d_out;               // quant | indices | loss

    // Workspace layout (~4.4 MB)
    char* ws = (char*)d_ws;
    unsigned short* chi   = (unsigned short*)(ws);                             // 2 MB
    unsigned short* clo   = (unsigned short*)(ws + (size_t)2 * KCODES * DIMV); // 2 MB
    float*          cnorm = (float*)(ws + (size_t)4 * KCODES * DIMV);          // 16 KB
    float*          xnorm = cnorm + KCODES;                                    // 64 KB
    int*            widx  = (int*)(xnorm + NROWS);                             // 64 KB
    float*          wmind = (float*)(widx + NROWS);                            // 64 KB

    prep_codebook<<<KCODES, 256, 0, stream>>>(cb, chi, clo, cnorm);
    prep_xnorm<<<NROWS / 256, 256, 0, stream>>>(x, xnorm);
    argmin_wmma<<<NROWS / 16 / 8, 256, 0, stream>>>(x, chi, clo, cnorm, xnorm,
                                                    widx, wmind);
    write_quant<<<(BATCH * DIMV * NSEQ) / 256, 256, 0, stream>>>(cb, widx, out);
    write_idx<<<NROWS / 256, 256, 0, stream>>>(
        widx, out + (size_t)BATCH * DIMV * NSEQ);
    loss_reduce<<<1, 256, 0, stream>>>(
        wmind, out + (size_t)BATCH * DIMV * NSEQ + NROWS);
}
